// RWKV_88983132438848
// MI455X (gfx1250) — matrix operand order repaired and checked by reading, not compiled, not run on hardware
//
#include <hip/hip_runtime.h>

// ---------------- problem constants ----------------
constexpr int B = 8, T = 2048, C = 2048, F = 8192;
constexpr size_t BT  = (size_t)B * T;          // 16384 rows
constexpr size_t BTC = BT * C;                 // 33.5M
constexpr size_t BTF = BT * F;                 // 134M
constexpr size_t BC  = (size_t)B * C;
constexpr size_t CC  = (size_t)C * C;
constexpr size_t FC  = (size_t)F * C;

typedef unsigned short u16;
typedef __attribute__((ext_vector_type(16))) __bf16 bf16x16;
typedef __attribute__((ext_vector_type(8)))  float  f32x8;

struct __align__(16) U4 { unsigned int a, b, c, d; };
union FragB16 { bf16x16 v; U4 q[2]; };

#if __has_builtin(__builtin_amdgcn_global_load_async_to_lds_b128) && \
    __has_builtin(__builtin_amdgcn_s_wait_asynccnt)
#define USE_ASYNC_LDS 1
// GCC-style int4 vector: matches the builtin's parameter type exactly
// ('__attribute__((__vector_size__(4*sizeof(int)))) int').
typedef int v4i __attribute__((vector_size(16)));
typedef v4i __attribute__((address_space(1)))* as1_v4i_p;   // global (HIP: __device__)
typedef v4i __attribute__((address_space(3)))* as3_v4i_p;   // LDS
#else
#define USE_ASYNC_LDS 0
#endif

__device__ __forceinline__ u16 f2bf(float f) {
  unsigned u = __builtin_bit_cast(unsigned, f);
  u += 0x7FFFu + ((u >> 16) & 1u);            // round-to-nearest-even
  return (u16)(u >> 16);
}

// ---------------- BF16 WMMA GEMM: out[M,N] = A[M,K] @ W[N,K]^T ----------------
// Block tile 128x64, K-step 32; 8 waves, each owns a 32x32 output tile
// (2 A-frags x 2 B-frags -> 4 WMMAs per K-step per wave).
// EPI 0: f32 out = acc
// EPI 1: f32 out = acc + Xadd   (residual)
// EPI 2: u16(bf16) out = bf16(relu(acc)^2)
#define LDT 48
template <int EPI>
__global__ __launch_bounds__(256) void gemm_bf16(
    const u16* __restrict__ A, const u16* __restrict__ W, void* __restrict__ outp,
    const float* __restrict__ Xadd, int M, int N, int K)
{
  __shared__ __align__(16) u16 ldsA[128 * LDT];
  __shared__ __align__(16) u16 ldsB[64 * LDT];

  const int tid  = threadIdx.x;
  const int wave = tid >> 5;
  const int lane = tid & 31;
  const int half = lane >> 4;     // 0..1
  const int l16  = lane & 15;

  const int mBlock = blockIdx.y * 128;
  const int nBlock = blockIdx.x * 64;
  const int mSub   = (wave & 3) * 32;   // wave's 32-row region
  const int nSub   = (wave >> 2) * 32;  // wave's 32-col region

  // staging: 16B chunks; A tile 128x32 -> 2 chunks/thread, B tile 64x32 -> 1
  const int ldRowA = tid >> 2;          // 0..63 (and +64 for chunk 1)
  const int ldCol  = (tid & 3) * 8;     // 0,8,16,24 (bf16 elems, 16B chunks)

  const int kA = half * 8;              // A-fragment K base (ISA 16-bit A layout)
  const int kB = half * 8;              // B-fragment K base (same layout as A)

  f32x8 c00 = {0.f,0.f,0.f,0.f,0.f,0.f,0.f,0.f};
  f32x8 c01 = {0.f,0.f,0.f,0.f,0.f,0.f,0.f,0.f};
  f32x8 c10 = {0.f,0.f,0.f,0.f,0.f,0.f,0.f,0.f};
  f32x8 c11 = {0.f,0.f,0.f,0.f,0.f,0.f,0.f,0.f};

  const u16* gA0 = A + (size_t)(mBlock + ldRowA)      * K + ldCol;
  const u16* gA1 = A + (size_t)(mBlock + ldRowA + 64) * K + ldCol;
  const u16* gW  = W + (size_t)(nBlock + ldRowA)      * K + ldCol;

  u16* sA0 = &ldsA[ldRowA * LDT + ldCol];
  u16* sA1 = &ldsA[(ldRowA + 64) * LDT + ldCol];
  u16* sB  = &ldsB[ldRowA * LDT + ldCol];

  for (int k0 = 0; k0 < K; k0 += 32) {
    if (k0 + 32 < K) {                  // pull next tile toward L2
      __builtin_prefetch(gA0 + k0 + 32, 0, 1);
      __builtin_prefetch(gA1 + k0 + 32, 0, 1);
      __builtin_prefetch(gW  + k0 + 32, 0, 1);
    }
#if USE_ASYNC_LDS
    __syncthreads();                    // previous iter's fragment reads done
    __builtin_amdgcn_global_load_async_to_lds_b128(
        (as1_v4i_p)(uintptr_t)(gA0 + k0),
        (as3_v4i_p)(unsigned)(uintptr_t)sA0, 0, 0);
    __builtin_amdgcn_global_load_async_to_lds_b128(
        (as1_v4i_p)(uintptr_t)(gA1 + k0),
        (as3_v4i_p)(unsigned)(uintptr_t)sA1, 0, 0);
    __builtin_amdgcn_global_load_async_to_lds_b128(
        (as1_v4i_p)(uintptr_t)(gW + k0),
        (as3_v4i_p)(unsigned)(uintptr_t)sB, 0, 0);
    __builtin_amdgcn_s_wait_asynccnt(0);
    __syncthreads();
#else
    U4 av0 = *(const U4*)(gA0 + k0);
    U4 av1 = *(const U4*)(gA1 + k0);
    U4 bv  = *(const U4*)(gW + k0);
    __syncthreads();                    // previous iter's fragment reads done
    *(U4*)sA0 = av0;
    *(U4*)sA1 = av1;
    *(U4*)sB  = bv;
    __syncthreads();
#endif

    FragB16 fa0, fa1, fb0, fb1;
    const u16* arow0 = &ldsA[(mSub + l16) * LDT];
    const u16* arow1 = &ldsA[(mSub + 16 + l16) * LDT];
    fa0.q[0] = *(const U4*)(arow0 + kA);          // K = kA..kA+7
    fa0.q[1] = *(const U4*)(arow0 + kA + 16);     // K = kA+16..kA+23
    fa1.q[0] = *(const U4*)(arow1 + kA);
    fa1.q[1] = *(const U4*)(arow1 + kA + 16);
    const u16* brow0 = &ldsB[(nSub + l16) * LDT];
    const u16* brow1 = &ldsB[(nSub + 16 + l16) * LDT];
    fb0.q[0] = *(const U4*)(brow0 + kB);          // K = kB..kB+7
    fb0.q[1] = *(const U4*)(brow0 + kB + 16);     // K = kB+16..kB+23
    fb1.q[0] = *(const U4*)(brow1 + kB);
    fb1.q[1] = *(const U4*)(brow1 + kB + 16);

    c00 = __builtin_amdgcn_wmma_f32_16x16x32_bf16(false, fa0.v, false, fb0.v,
                                                  (short)0, c00, false, false);
    c01 = __builtin_amdgcn_wmma_f32_16x16x32_bf16(false, fa0.v, false, fb1.v,
                                                  (short)0, c01, false, false);
    c10 = __builtin_amdgcn_wmma_f32_16x16x32_bf16(false, fa1.v, false, fb0.v,
                                                  (short)0, c10, false, false);
    c11 = __builtin_amdgcn_wmma_f32_16x16x32_bf16(false, fa1.v, false, fb1.v,
                                                  (short)0, c11, false, false);
  }

  // C/D layout: VGPR r -> M = r + 8*half ; N = lane&15
  const int n0 = nBlock + nSub + l16;
#pragma unroll
  for (int r = 0; r < 8; ++r) {
    const int m0 = mBlock + mSub + r + half * 8;  // fa0 tile rows
    const int m1 = m0 + 16;                       // fa1 tile rows
    const size_t i00 = (size_t)m0 * N + n0;
    const size_t i01 = i00 + 16;
    const size_t i10 = (size_t)m1 * N + n0;
    const size_t i11 = i10 + 16;
    float v00 = c00[r], v01 = c01[r], v10 = c10[r], v11 = c11[r];
    if constexpr (EPI == 0) {
      ((float*)outp)[i00] = v00; ((float*)outp)[i01] = v01;
      ((float*)outp)[i10] = v10; ((float*)outp)[i11] = v11;
    } else if constexpr (EPI == 1) {
      ((float*)outp)[i00] = v00 + Xadd[i00]; ((float*)outp)[i01] = v01 + Xadd[i01];
      ((float*)outp)[i10] = v10 + Xadd[i10]; ((float*)outp)[i11] = v11 + Xadd[i11];
    } else {
      v00 = fmaxf(v00, 0.f); v01 = fmaxf(v01, 0.f);
      v10 = fmaxf(v10, 0.f); v11 = fmaxf(v11, 0.f);
      ((u16*)outp)[i00] = f2bf(v00 * v00); ((u16*)outp)[i01] = f2bf(v01 * v01);
      ((u16*)outp)[i10] = f2bf(v10 * v10); ((u16*)outp)[i11] = f2bf(v11 * v11);
    }
  }
}

// ---------------- LayerNorm: one block per (b,t) row ----------------
__global__ __launch_bounds__(256) void ln_kernel(
    const float* __restrict__ x, const float* __restrict__ g,
    const float* __restrict__ bb, float* __restrict__ out)
{
  __shared__ float red[256];
  const size_t row = blockIdx.x;
  const float* xr = x + row * C;
  float s = 0.f, ss = 0.f;
  for (int i = threadIdx.x; i < C; i += 256) { float v = xr[i]; s += v; ss += v * v; }
  red[threadIdx.x] = s; __syncthreads();
  for (int o = 128; o > 0; o >>= 1) { if (threadIdx.x < o) red[threadIdx.x] += red[threadIdx.x + o]; __syncthreads(); }
  const float mean = red[0] * (1.0f / C); __syncthreads();
  red[threadIdx.x] = ss; __syncthreads();
  for (int o = 128; o > 0; o >>= 1) { if (threadIdx.x < o) red[threadIdx.x] += red[threadIdx.x + o]; __syncthreads(); }
  const float var  = red[0] * (1.0f / C) - mean * mean;
  const float rstd = rsqrtf(var + 1e-5f);
  float* orow = out + row * C;
  for (int i = threadIdx.x; i < C; i += 256)
    orow[i] = (xr[i] - mean) * rstd * g[i] + bb[i];
}

// ---------------- WKV recurrence: one thread per (b,c) ----------------
__global__ __launch_bounds__(256) void wkv_kernel(
    const float* __restrict__ k, const float* __restrict__ v,
    const float* __restrict__ td, const float* __restrict__ tf,
    const float* __restrict__ st_in, float* __restrict__ y, float* __restrict__ st_out)
{
  const size_t idx = (size_t)blockIdx.x * 256 + threadIdx.x;
  if (idx >= BC) return;
  const int b = (int)(idx / C), c = (int)(idx % C);
  const float w = -expf(td[c]);
  const float u = tf[c];
  const float* si = st_in + idx * 3;
  float aa = si[0], bb = si[1], pp = si[2];
  const size_t base = (size_t)b * T * C + c;
  for (int t = 0; t < T; ++t) {
    const size_t o = base + (size_t)t * C;
    const float kt = k[o], vt = v[o];
    const float ww = u + kt;
    const float p  = fmaxf(pp, ww);
    const float e1 = expf(pp - p), e2 = expf(ww - p);
    y[o] = (e1 * aa + e2 * vt) / (e1 * bb + e2);
    const float ww2 = pp + w;
    const float p2  = fmaxf(ww2, kt);
    const float e1b = expf(ww2 - p2), e2b = expf(kt - p2);
    aa = e1b * aa + e2b * vt;
    bb = e1b * bb + e2b;
    pp = p2;
  }
  float* so = st_out + idx * 3;
  so[0] = aa; so[1] = bb; so[2] = pp;
}

// ---------------- elementwise helpers ----------------
__global__ void cvt_bf16(const float* __restrict__ s, u16* __restrict__ d, size_t n) {
  size_t i = (size_t)blockIdx.x * 256 + threadIdx.x;
  if (i < n) d[i] = f2bf(s[i]);
}
__global__ void mix3_kernel(const float* __restrict__ xa, const float* __restrict__ shift,
                            const float* __restrict__ tk, const float* __restrict__ tv,
                            const float* __restrict__ tr,
                            u16* __restrict__ mk, u16* __restrict__ mv, u16* __restrict__ mr) {
  size_t i = (size_t)blockIdx.x * 256 + threadIdx.x;
  if (i >= BTC) return;
  const int c = (int)(i % C);
  const size_t bt = i / C;
  const int t = (int)(bt % T);
  const float cur  = xa[i];
  const float prev = (t == 0) ? shift[(bt / T) * C + c] : xa[i - C];
  float m;
  m = tk[c]; mk[i] = f2bf(cur * m + prev * (1.f - m));
  m = tv[c]; mv[i] = f2bf(cur * m + prev * (1.f - m));
  m = tr[c]; mr[i] = f2bf(cur * m + prev * (1.f - m));
}
__global__ void mix2_kernel(const float* __restrict__ xf, const float* __restrict__ shift,
                            const float* __restrict__ tk, const float* __restrict__ tr,
                            u16* __restrict__ fk, u16* __restrict__ fr) {
  size_t i = (size_t)blockIdx.x * 256 + threadIdx.x;
  if (i >= BTC) return;
  const int c = (int)(i % C);
  const size_t bt = i / C;
  const int t = (int)(bt % T);
  const float cur  = xf[i];
  const float prev = (t == 0) ? shift[(bt / T) * C + c] : xf[i - C];
  float m;
  m = tk[c]; fk[i] = f2bf(cur * m + prev * (1.f - m));
  m = tr[c]; fr[i] = f2bf(cur * m + prev * (1.f - m));
}
__global__ void sigmul_kernel(const float* __restrict__ r, const float* __restrict__ y,
                              u16* __restrict__ z, size_t n) {
  size_t i = (size_t)blockIdx.x * 256 + threadIdx.x;
  if (i < n) z[i] = f2bf((1.0f / (1.0f + expf(-r[i]))) * y[i]);
}
__global__ void final_kernel(float* __restrict__ x, const float* __restrict__ rf,
                             const float* __restrict__ kv, size_t n) {
  size_t i = (size_t)blockIdx.x * 256 + threadIdx.x;
  if (i < n) x[i] = x[i] + (1.0f / (1.0f + expf(-rf[i]))) * kv[i];
}
__global__ void last_row_kernel(const float* __restrict__ src, float* __restrict__ dst) {
  size_t i = (size_t)blockIdx.x * 256 + threadIdx.x;
  if (i >= BC) return;
  const size_t b = i / C, c = i % C;
  dst[i] = src[(b * T + (T - 1)) * C + c];
}

// ---------------- launch ----------------
extern "C" void kernel_launch(void* const* d_in, const int* in_sizes, int n_in,
                              void* d_out, int out_size, void* d_ws, size_t ws_size,
                              hipStream_t stream)
{
  const float* x          = (const float*)d_in[0];
  const float* shift_att  = (const float*)d_in[1];
  const float* shift_ffn  = (const float*)d_in[2];
  const float* wkv_state  = (const float*)d_in[3];
  const float* ln1_g = (const float*)d_in[4];
  const float* ln1_b = (const float*)d_in[5];
  const float* ln2_g = (const float*)d_in[6];
  const float* ln2_b = (const float*)d_in[7];
  const float* tmk = (const float*)d_in[8];
  const float* tmv = (const float*)d_in[9];
  const float* tmr = (const float*)d_in[10];
  const float* time_decay = (const float*)d_in[11];
  const float* time_first = (const float*)d_in[12];
  const float* Wk  = (const float*)d_in[13];
  const float* Wv  = (const float*)d_in[14];
  const float* Wr  = (const float*)d_in[15];
  const float* Wo  = (const float*)d_in[16];
  const float* fmk = (const float*)d_in[17];
  const float* fmr = (const float*)d_in[18];
  const float* Wkf = (const float*)d_in[19];
  const float* Wvf = (const float*)d_in[20];
  const float* Wrf = (const float*)d_in[21];

  float* out_x       = (float*)d_out;
  float* out_xa_last = out_x + BTC;
  float* out_xf_last = out_xa_last + BC;
  float* out_state   = out_xf_last + BC;

  // ---- workspace carve (256B aligned slabs) ----
  uintptr_t base = (uintptr_t)d_ws;
  size_t cur = 0;
  auto take = [&](size_t bytes) -> void* {
    void* p = (void*)(base + cur);
    cur += (bytes + 255) & ~(size_t)255;
    return p;
  };
  u16* Wkb  = (u16*)take(CC * 2);
  u16* Wvb  = (u16*)take(CC * 2);
  u16* Wrb  = (u16*)take(CC * 2);
  u16* Wob  = (u16*)take(CC * 2);
  u16* Wrfb = (u16*)take(CC * 2);
  u16* Wkfb = (u16*)take(FC * 2);
  u16* Wvfb = (u16*)take(FC * 2);
  float* xa   = (float*)take(BTC * 4);   // reused as xf
  u16*   mk   = (u16*)take(BTC * 2);     // reused as fk
  u16*   mv   = (u16*)take(BTC * 2);
  u16*   mr   = (u16*)take(BTC * 2);     // reused as fr
  float* kbuf = (float*)take(BTC * 4);
  float* vbuf = (float*)take(BTC * 4);
  float* rbuf = (float*)take(BTC * 4);   // reused as rf (pre-sigmoid)
  float* ybuf = (float*)take(BTC * 4);
  u16*   zbf  = (u16*)take(BTC * 2);
  u16*   kfb  = (u16*)take(BTF * 2);
  float* kvb  = (float*)take(BTC * 4);

  auto nb = [](size_t n) { return (unsigned)((n + 255) / 256); };

  // 1) weights -> bf16
  cvt_bf16<<<nb(CC), 256, 0, stream>>>(Wk,  Wkb,  CC);
  cvt_bf16<<<nb(CC), 256, 0, stream>>>(Wv,  Wvb,  CC);
  cvt_bf16<<<nb(CC), 256, 0, stream>>>(Wr,  Wrb,  CC);
  cvt_bf16<<<nb(CC), 256, 0, stream>>>(Wo,  Wob,  CC);
  cvt_bf16<<<nb(CC), 256, 0, stream>>>(Wrf, Wrfb, CC);
  cvt_bf16<<<nb(FC), 256, 0, stream>>>(Wkf, Wkfb, FC);
  cvt_bf16<<<nb(FC), 256, 0, stream>>>(Wvf, Wvfb, FC);

  // 2) LN1 + shift/mix
  ln_kernel<<<(unsigned)BT, 256, 0, stream>>>(x, ln1_g, ln1_b, xa);
  last_row_kernel<<<nb(BC), 256, 0, stream>>>(xa, out_xa_last);
  mix3_kernel<<<nb(BTC), 256, 0, stream>>>(xa, shift_att, tmk, tmv, tmr, mk, mv, mr);

  // 3) k/v/r projections (WMMA bf16, fp32 accum)
  dim3 gC(C / 64, (unsigned)(BT / 128));
  gemm_bf16<0><<<gC, 256, 0, stream>>>(mk, Wkb, kbuf, nullptr, (int)BT, C, C);
  gemm_bf16<0><<<gC, 256, 0, stream>>>(mv, Wvb, vbuf, nullptr, (int)BT, C, C);
  gemm_bf16<0><<<gC, 256, 0, stream>>>(mr, Wrb, rbuf, nullptr, (int)BT, C, C);

  // 4) WKV scan (fp32, writes new state straight into d_out)
  wkv_kernel<<<nb(BC), 256, 0, stream>>>(kbuf, vbuf, time_decay, time_first,
                                         wkv_state, ybuf, out_state);

  // 5) sigmoid(r)*y -> bf16, then output projection with residual add
  sigmul_kernel<<<nb(BTC), 256, 0, stream>>>(rbuf, ybuf, zbf, BTC);
  gemm_bf16<1><<<gC, 256, 0, stream>>>(zbf, Wob, out_x, x, (int)BT, C, C);

  // 6) LN2 + shift/mix (channel-mix)
  ln_kernel<<<(unsigned)BT, 256, 0, stream>>>(out_x, ln2_g, ln2_b, xa);
  last_row_kernel<<<nb(BC), 256, 0, stream>>>(xa, out_xf_last);
  mix2_kernel<<<nb(BTC), 256, 0, stream>>>(xa, shift_ffn, fmk, fmr, mk, mr);

  // 7) FFN: kf = relu(fk@Wkf^T)^2 (bf16 epilogue), kv = kf@Wvf^T, rf = fr@Wrf^T
  dim3 gF(F / 64, (unsigned)(BT / 128));
  gemm_bf16<2><<<gF, 256, 0, stream>>>(mk, Wkfb, kfb, nullptr, (int)BT, F, C);
  gemm_bf16<0><<<gC, 256, 0, stream>>>(kfb, Wvfb, kvb, nullptr, (int)BT, C, F);
  gemm_bf16<0><<<gC, 256, 0, stream>>>(mr, Wrfb, rbuf, nullptr, (int)BT, C, C);

  // 8) x += sigmoid(rf) * kv
  final_kernel<<<nb(BTC), 256, 0, stream>>>(out_x, rbuf, kvb, BTC);
}
